// AttentionModule_6373731467427
// MI455X (gfx1250) — matrix operand order repaired and checked by reading, not compiled, not run on hardware
//
#include <hip/hip_runtime.h>
#include <math.h>

// ---------- problem constants ----------
#define CIN   256
#define CF    32
#define HIMG  128
#define WIMG  128
#define HW    (HIMG*WIMG)     // 16384
#define BATCH 4
#define WS    8
#define STRIDE 4
#define NWIN  32              // H/STRIDE

typedef __bf16 bf16;
typedef __attribute__((ext_vector_type(16))) __bf16 v16bf;
typedef __attribute__((ext_vector_type(8)))  float  v8f;

union Frag   { v16bf v; uint4 q[2]; };
union Pack4  { bf16 h[4]; unsigned long long u; };
union Pack8  { bf16 h[8]; uint4 q4; };

// ---------- WMMA fragment loaders (LDS, bf16, layouts per ISA 7.12.2) ----------
// A matrix 16x32 (MxK): lane half h: elements e<8 -> K = koff + h*8 + e
//                                    e>=8 -> K = koff + 16 + h*8 + (e-8)
__device__ __forceinline__ v16bf load_frag_A(const bf16* base, int ldk, int koff, int lane) {
  const int r = lane & 15, half = lane >> 4;
  const bf16* p = base + (size_t)r * ldk + koff;
  Frag f;
  f.q[0] = *(const uint4*)(p + half * 8);
  f.q[1] = *(const uint4*)(p + 16 + half * 8);
  return f.v;
}
// B matrix 32x16 (KxN), stored in LDS as [N][K]: lane half h: e<8 -> K = koff + h*8 + e, e>=8 -> K = koff + 16 + h*8 + (e-8)
__device__ __forceinline__ v16bf load_frag_B(const bf16* base, int ldk, int koff, int lane) {
  const int n = lane & 15, half = lane >> 4;
  const bf16* p = base + (size_t)n * ldk + koff;
  Frag f;
  f.q[0] = *(const uint4*)(p + half * 8);
  f.q[1] = *(const uint4*)(p + 16 + half * 8);
  return f.v;
}

__device__ __forceinline__ v8f wmma_bf16(v16bf a, v16bf b, v8f c) {
  return __builtin_amdgcn_wmma_f32_16x16x32_bf16(false, a, false, b, (short)0, c, false, false);
}

// ============================================================================
// Kernel 1: fused 1x1 convs as GEMMs  Y[m,p] = sum_k W[m,k] * X[k,p] + bias[m]
//   blockIdx.y: 0 -> f (Wf @ f_g), 1 -> g (Wg @ f_l), 2..9 -> h slices (Wh @ f_l)
//   tile: M=32 x N=64 per block (8 waves: 2 M-subtiles x 4 N-subtiles), K-step 32
//   outputs stored as bf16 in workspace
// ============================================================================
__global__ __launch_bounds__(256) void conv1x1_wmma_kernel(
    const float* __restrict__ f_g, const float* __restrict__ f_l,
    const float* __restrict__ Wf,  const float* __restrict__ bf_,
    const float* __restrict__ Wg,  const float* __restrict__ bg_,
    const float* __restrict__ Wh,  const float* __restrict__ bh_,
    unsigned short* __restrict__ wsf_u, unsigned short* __restrict__ wsg_u,
    unsigned short* __restrict__ wsh_u)
{
  __shared__ __align__(16) bf16 Xs[64 * 32];   // [N=64][K=32]
  __shared__ __align__(16) bf16 As[32 * 32];   // [M=32][K=32]

  const int b   = blockIdx.z;
  const int yid = blockIdx.y;
  const float* in; const float* Wt; const float* bias; bf16* out; int m0; int Cout;
  if (yid == 0)      { in = f_g; Wt = Wf; bias = bf_; out = (bf16*)wsf_u; m0 = 0;            Cout = CF;  }
  else if (yid == 1) { in = f_l; Wt = Wg; bias = bg_; out = (bf16*)wsg_u; m0 = 0;            Cout = CF;  }
  else               { in = f_l; Wt = Wh; bias = bh_; out = (bf16*)wsh_u; m0 = (yid-2)*32;   Cout = CIN; }

  const int p0   = blockIdx.x * 64;
  const int tid  = threadIdx.x;
  const int lane = tid & 31;
  const int wave = tid >> 5;
  const int mw   = wave >> 2;        // 0..1
  const int nw   = wave & 3;         // 0..3
  const int half = lane >> 4;

  // staging indices (hoisted)
  const int xp = tid & 63;           // pixel within tile
  const int kg = tid >> 6;           // K-group (0..3) of 8 channels
  const int mr = tid >> 3;           // weight row 0..31
  const int kc = (tid & 7) * 4;      // weight cols (4 per thread)

  v8f acc = {};

  for (int kk = 0; kk < CIN / 32; ++kk) {
    const int k0 = kk * 32;
    // stage X tile directly in [N][K] layout:
    //   thread gathers 8 K-consecutive channels of one pixel (coalesced across
    //   lanes), packs to bf16, lands with a single 16B LDS store.
    {
      const float* src = in + ((size_t)b * CIN + k0 + kg * 8) * HW + p0 + xp;
      Pack8 pk;
      #pragma unroll
      for (int e = 0; e < 8; ++e) pk.h[e] = (bf16)src[(size_t)e * HW];
      *(uint4*)&Xs[xp * 32 + kg * 8] = pk.q4;
    }
    // stage weight tile: W[m0+mr][k0+kc..kc+3] -> As[mr][kc..]
    {
      const float* src = Wt + (size_t)(m0 + mr) * CIN + k0 + kc;
      float4 v = *(const float4*)src;
      Pack4 p; p.h[0] = (bf16)v.x; p.h[1] = (bf16)v.y; p.h[2] = (bf16)v.z; p.h[3] = (bf16)v.w;
      *(unsigned long long*)&As[mr * 32 + kc] = p.u;
    }
    __syncthreads();
    v16bf aF = load_frag_A(&As[mw * 16 * 32], 32, 0, lane);
    v16bf bF = load_frag_B(&Xs[nw * 16 * 32], 32, 0, lane);
    acc = wmma_bf16(aF, bF, acc);
    __syncthreads();
  }

  // store D (16x16 f32 layout: VGPR r -> M = r + half*8, lanes&15 -> N)
  const int p = p0 + nw * 16 + (lane & 15);
  #pragma unroll
  for (int r = 0; r < 8; ++r) {
    const int o = m0 + mw * 16 + r + half * 8;
    const float v = acc[r] + bias[o];
    out[((size_t)b * Cout + o) * HW + p] = (bf16)v;
  }
}

// ---------- guarded 8-wide bf16 row load (handles right/bottom window pad) ----
__device__ __forceinline__ void load_row8(const bf16* base, int y, int x0, bf16* vals) {
  if (y < HIMG) {
    const bf16* p = base + (size_t)y * WIMG + x0;
    Pack4 a; a.u = *(const unsigned long long*)p;          // x0 is multiple of 4 -> 8B aligned
    #pragma unroll
    for (int e = 0; e < 4; ++e) vals[e] = a.h[e];
    if (x0 + 8 <= WIMG) {
      Pack4 c; c.u = *(const unsigned long long*)(p + 4);
      #pragma unroll
      for (int e = 0; e < 4; ++e) vals[4 + e] = c.h[e];
    } else {
      #pragma unroll
      for (int e = 0; e < 4; ++e) vals[4 + e] = (bf16)0.0f;
    }
  } else {
    #pragma unroll
    for (int e = 0; e < 8; ++e) vals[e] = (bf16)0.0f;
  }
}

// ============================================================================
// Kernel 2: per-window attention. One 128-thread block (4 waves) per window.
//   Only the 16 query positions (wy<4, wx<4) are computed — they are the only
//   ones the reference keeps. Keys span the full (masked) 8x8 window.
// ============================================================================
__global__ __launch_bounds__(128) void window_attn_wmma_kernel(
    const unsigned short* __restrict__ wsf_u, const unsigned short* __restrict__ wsg_u,
    const unsigned short* __restrict__ wsh_u, float* __restrict__ out)
{
  __shared__ __align__(16) bf16 fA[16 * 32];     // [q][c]   A of logits GEMM
  __shared__ __align__(16) bf16 gB[64 * 32];     // [m][c]   B of logits GEMM ([N][K])
  __shared__ __align__(16) bf16 hA[256 * 64];    // [c][m]   A of output GEMM
  __shared__ __align__(16) bf16 attB[16 * 64];   // [q][m]   B of output GEMM ([N][K])
  __shared__ float lg[16 * 64];                  // logits scratch

  const bf16* ws_f = (const bf16*)wsf_u;
  const bf16* ws_g = (const bf16*)wsg_u;
  const bf16* ws_h = (const bf16*)wsh_u;

  const int j = blockIdx.x, i = blockIdx.y, b = blockIdx.z;
  const int x0 = j * STRIDE, y0 = i * STRIDE;
  const int tid = threadIdx.x, lane = tid & 31, wave = tid >> 5;
  const int half = lane >> 4, n = lane & 15;

  // ---- stage fw (16 valid queries) : fA[q=wy*4+wx][c] ----
  {
    const int c = tid >> 2, wy = tid & 3;     // 32*4 = 128 -> one per thread
    const bf16* src = ws_f + ((size_t)b * CF + c) * HW + (size_t)(y0 + wy) * WIMG + x0;
    Pack4 p; p.u = *(const unsigned long long*)src;
    #pragma unroll
    for (int wx = 0; wx < 4; ++wx) fA[(wy * 4 + wx) * 32 + c] = p.h[wx];
  }
  // ---- stage gw : gB[m=my*8+mx][c], zero-filled outside image ----
  #pragma unroll
  for (int it = 0; it < 2; ++it) {
    const int idx = tid + it * 128;           // 0..255
    const int c = idx >> 3, my = idx & 7;
    bf16 vals[8];
    load_row8(ws_g + ((size_t)b * CF + c) * HW, y0 + my, x0, vals);
    #pragma unroll
    for (int mx = 0; mx < 8; ++mx) gB[(my * 8 + mx) * 32 + c] = vals[mx];
  }
  // ---- stage hw : hA[c][m], contiguous in m -> one 16B LDS store per row ----
  #pragma unroll
  for (int it = 0; it < 16; ++it) {
    const int idx = tid + it * 128;           // 0..2047
    const int c = idx >> 3, my = idx & 7;
    Pack8 pk;
    load_row8(ws_h + ((size_t)b * CIN + c) * HW, y0 + my, x0, pk.h);
    *(uint4*)&hA[c * 64 + my * 8] = pk.q4;
  }
  __syncthreads();

  // ---- logits: 16(q) x 64(m), K = Cf = 32. One 16x16 N-tile per wave. ----
  {
    v16bf aF = load_frag_A(fA, 32, 0, lane);
    v16bf bF = load_frag_B(&gB[wave * 16 * 32], 32, 0, lane);
    v8f c = {};
    c = wmma_bf16(aF, bF, c);
    #pragma unroll
    for (int r = 0; r < 8; ++r) lg[(r + half * 8) * 64 + wave * 16 + n] = c[r];
  }
  __syncthreads();

  // ---- masked softmax over m (f32), 4 lanes per query row, shfl combine ----
  if (tid < 64) {
    const int q = tid >> 2, sub = tid & 3;
    const float* row = &lg[q * 64];
    float v[16];
    float pmax = -1e30f;
    #pragma unroll
    for (int e = 0; e < 16; ++e) {
      const int m = sub * 16 + e;
      const bool valid = ((y0 + (m >> 3)) < HIMG) && ((x0 + (m & 7)) < WIMG);
      const float t = valid ? row[m] : -1.0e9f;
      v[e] = t;
      pmax = fmaxf(pmax, t);
    }
    pmax = fmaxf(pmax, __shfl_xor(pmax, 1));
    pmax = fmaxf(pmax, __shfl_xor(pmax, 2));
    float ps = 0.0f;
    #pragma unroll
    for (int e = 0; e < 16; ++e) { const float ev = expf(v[e] - pmax); v[e] = ev; ps += ev; }
    ps += __shfl_xor(ps, 1);
    ps += __shfl_xor(ps, 2);
    const float inv = 1.0f / ps;
    Pack8 lo, hi;
    #pragma unroll
    for (int e = 0; e < 8; ++e) { lo.h[e] = (bf16)(v[e] * inv); hi.h[e] = (bf16)(v[8 + e] * inv); }
    *(uint4*)&attB[q * 64 + sub * 16]     = lo.q4;
    *(uint4*)&attB[q * 64 + sub * 16 + 8] = hi.q4;
  }
  __syncthreads();

  // ---- output GEMM: 256(c) x 16(q), K = 64(m). 4 M-tiles per wave. ----
  v16bf bF0 = load_frag_B(attB, 64, 0,  lane);
  v16bf bF1 = load_frag_B(attB, 64, 32, lane);
  const int q = n;
  const int y = y0 + (q >> 2);
  const int x = x0 + (q & 3);
  #pragma unroll
  for (int t = 0; t < 4; ++t) {
    const int c0 = (wave * 4 + t) * 16;
    v8f acc = {};
    v16bf a0 = load_frag_A(&hA[(size_t)c0 * 64], 64, 0,  lane);
    acc = wmma_bf16(a0, bF0, acc);
    v16bf a1 = load_frag_A(&hA[(size_t)c0 * 64], 64, 32, lane);
    acc = wmma_bf16(a1, bF1, acc);
    #pragma unroll
    for (int r = 0; r < 8; ++r) {
      const int c = c0 + r + half * 8;
      out[((size_t)b * CIN + c) * HW + (size_t)y * WIMG + x] = acc[r];
    }
  }
}

// ============================================================================
extern "C" void kernel_launch(void* const* d_in, const int* in_sizes, int n_in,
                              void* d_out, int out_size, void* d_ws, size_t ws_size,
                              hipStream_t stream) {
  const float* f_g = (const float*)d_in[0];
  const float* f_l = (const float*)d_in[1];
  const float* Wf  = (const float*)d_in[2];
  const float* bf_ = (const float*)d_in[3];
  const float* Wg  = (const float*)d_in[4];
  const float* bg_ = (const float*)d_in[5];
  const float* Wh  = (const float*)d_in[6];
  const float* bh_ = (const float*)d_in[7];
  float* out = (float*)d_out;

  // workspace layout (bf16): f (4MB) | g (4MB) | h (32MB)
  unsigned short* wsf = (unsigned short*)d_ws;
  unsigned short* wsg = wsf + (size_t)BATCH * CF  * HW;
  unsigned short* wsh = wsg + (size_t)BATCH * CF  * HW;

  // conv: grid.x = pixel tiles (HW/64), grid.y = {f, g, h0..h7}, grid.z = batch
  conv1x1_wmma_kernel<<<dim3(HW / 64, 2 + CIN / 32, BATCH), 256, 0, stream>>>(
      f_g, f_l, Wf, bf_, Wg, bg_, Wh, bh_, wsf, wsg, wsh);

  // attention: one block per (j, i, b) window
  window_attn_wmma_kernel<<<dim3(NWIN, NWIN, BATCH), 128, 0, stream>>>(
      wsf, wsg, wsh, out);
}
